// CoAttention_2534030705334
// MI455X (gfx1250) — matrix operand order repaired and checked by reading, not compiled, not run on hardware
//
#include <hip/hip_runtime.h>
#include <hip/hip_bf16.h>

#define DEV __device__ __forceinline__

typedef __attribute__((ext_vector_type(16))) __bf16 v16bf;
typedef __attribute__((ext_vector_type(8)))  float  v8f;

// round-to-nearest-even f32 -> bf16
DEV __bf16 f2bf(float f) {
  unsigned u = __float_as_uint(f);
  unsigned r = (u + 0x7FFFu + ((u >> 16) & 1u)) >> 16;
  unsigned short s = (unsigned short)r;
  return __builtin_bit_cast(__bf16, s);
}

DEV v8f wmma_bf16(v16bf a, v16bf b, v8f c) {
  // (neg_a, A, neg_b, B, c_mod, C, reuse_a, reuse_b)
  return __builtin_amdgcn_wmma_f32_16x16x32_bf16(false, a, false, b, (short)0, c,
                                                 false, false);
}

// ---------------- WMMA fragment loaders (wave32, 16x16x32 bf16) ----------------
// A operand (16x32, M x K): lane = half*16 + idx (idx = M row);
// element e -> K = half*16 + e; two contiguous 16B chunks.
DEV v16bf frag_a_row(const __bf16* base, int ld) {  // base = &src[row0][k0], row-major
  int lane = threadIdx.x & 31;
  int idx = lane & 15, half = lane >> 4;
  const __bf16* p = base + (size_t)idx * ld + half * 16;
  v16bf r;
#pragma unroll
  for (int e = 0; e < 8; ++e) r[e] = p[e];
#pragma unroll
  for (int e = 0; e < 8; ++e) r[8 + e] = p[8 + e];
  return r;
}

// B operand (32x16, K x N): lane = half*16 + idx (idx = N col); element e -> K = half*16 + e
// B = src^T where src is row-major [n][k] stride ld: one contiguous 32B chunk per lane.
DEV v16bf frag_b_rowT(const __bf16* base, int ld) {  // base = &src[n0][k0]
  int lane = threadIdx.x & 31;
  int idx = lane & 15, half = lane >> 4;
  const __bf16* p = base + (size_t)idx * ld + half * 16;
  v16bf r;
#pragma unroll
  for (int e = 0; e < 16; ++e) r[e] = p[e];
  return r;
}

// ---------------- 32x64 per-wave WMMA tile core ----------------
struct Tile32x64 { v8f c[2][4]; };

DEV void mma_loop(const __bf16* __restrict__ Abase, int lda,
                  const __bf16* __restrict__ Bbase, int ldb,
                  int Kred, Tile32x64& t) {
  for (int kk = 0; kk < Kred; kk += 32) {
    if (kk + 32 < Kred) {
      __builtin_prefetch(Abase + kk + 32, 0, 3);
      __builtin_prefetch(Bbase + kk + 32, 0, 3);
    }
    v16bf a0 = frag_a_row(Abase + kk, lda);
    v16bf a1 = frag_a_row(Abase + (size_t)16 * lda + kk, lda);
    v16bf b0 = frag_b_rowT(Bbase + kk, ldb);
    v16bf b1 = frag_b_rowT(Bbase + (size_t)16 * ldb + kk, ldb);
    v16bf b2 = frag_b_rowT(Bbase + (size_t)32 * ldb + kk, ldb);
    v16bf b3 = frag_b_rowT(Bbase + (size_t)48 * ldb + kk, ldb);
    t.c[0][0] = wmma_bf16(a0, b0, t.c[0][0]);
    t.c[0][1] = wmma_bf16(a0, b1, t.c[0][1]);
    t.c[0][2] = wmma_bf16(a0, b2, t.c[0][2]);
    t.c[0][3] = wmma_bf16(a0, b3, t.c[0][3]);
    t.c[1][0] = wmma_bf16(a1, b0, t.c[1][0]);
    t.c[1][1] = wmma_bf16(a1, b1, t.c[1][1]);
    t.c[1][2] = wmma_bf16(a1, b2, t.c[1][2]);
    t.c[1][3] = wmma_bf16(a1, b3, t.c[1][3]);
  }
}

// ---------------- elementwise convert f32 -> bf16 ----------------
__global__ __launch_bounds__(256) void k_cvt(const float* __restrict__ in,
                                             __bf16* __restrict__ out, size_t n) {
  size_t i = ((size_t)blockIdx.x * 256 + threadIdx.x) * 4;
  if (i + 3 < n) {
    float4 v = *(const float4*)(in + i);
    out[i + 0] = f2bf(v.x);
    out[i + 1] = f2bf(v.y);
    out[i + 2] = f2bf(v.z);
    out[i + 3] = f2bf(v.w);
  }
}

// -------- per-batch transpose + convert: fp32 [n][R][C] -> bf16 [n][C][R] --------
__global__ __launch_bounds__(256) void k_cvt_t(const float* __restrict__ in,
                                               __bf16* __restrict__ out,
                                               int R, int C) {
  __shared__ __bf16 tile[32][33];
  int n = blockIdx.z;
  const float* src = in + (size_t)n * R * C;
  __bf16* dst = out + (size_t)n * R * C;
  int tx = threadIdx.x & 31, ty = threadIdx.x >> 5;  // 32x8
  int r0 = blockIdx.y * 32, c0 = blockIdx.x * 32;
#pragma unroll
  for (int j = 0; j < 32; j += 8)
    tile[ty + j][tx] = f2bf(src[(size_t)(r0 + ty + j) * C + (c0 + tx)]);
  __syncthreads();
#pragma unroll
  for (int j = 0; j < 32; j += 8)
    dst[(size_t)(c0 + ty + j) * R + (r0 + tx)] = tile[tx][ty + j];
}

// ---------------- a2p = a2 @ Gw^T + b  (rows = N*K, cols = D, reduce D) ------
__global__ __launch_bounds__(256) void k_linear(const __bf16* __restrict__ A,
                                                const __bf16* __restrict__ W,
                                                const float* __restrict__ bias,
                                                __bf16* __restrict__ out, int D) {
  int wid = threadIdx.x >> 5;
  int row0 = blockIdx.y * 64 + (wid & 1) * 32;
  int col0 = blockIdx.x * 256 + (wid >> 1) * 64;
  Tile32x64 t = {};
  mma_loop(A + (size_t)row0 * D, D, W + (size_t)col0 * D, D, D, t);
  int lane = threadIdx.x & 31, idx = lane & 15, half = lane >> 4;
#pragma unroll
  for (int i = 0; i < 2; ++i)
#pragma unroll
    for (int j = 0; j < 4; ++j) {
      int c = col0 + j * 16 + idx;
      float bv = bias[c];
#pragma unroll
      for (int v = 0; v < 8; ++v) {
        int rr = row0 + i * 16 + half * 8 + v;
        out[(size_t)rr * D + c] = f2bf(t.c[i][j][v] + bv);
      }
    }
}

// ------- generic batched GEMM, both operands contiguous pattern, f32 out -------
__global__ __launch_bounds__(256) void k_gemm_f32(
    const __bf16* __restrict__ A, int lda, size_t sA,
    const __bf16* __restrict__ B, int ldb, size_t sB,
    float* __restrict__ C, int ldc, size_t sC, int Kred) {
  int n = blockIdx.z;
  const __bf16* An = A + (size_t)n * sA;
  const __bf16* Bn = B + (size_t)n * sB;
  float* Cn = C + (size_t)n * sC;
  int wid = threadIdx.x >> 5;
  int row0 = blockIdx.y * 64 + (wid & 1) * 32;
  int col0 = blockIdx.x * 256 + (wid >> 1) * 64;
  Tile32x64 t = {};
  mma_loop(An + (size_t)row0 * lda, lda, Bn + (size_t)col0 * ldb, ldb, Kred, t);
  int lane = threadIdx.x & 31, idx = lane & 15, half = lane >> 4;
#pragma unroll
  for (int i = 0; i < 2; ++i)
#pragma unroll
    for (int j = 0; j < 4; ++j) {
      int c = col0 + j * 16 + idx;
#pragma unroll
      for (int v = 0; v < 8; ++v) {
        int rr = row0 + i * 16 + half * 8 + v;
        Cn[(size_t)rr * ldc + c] = t.c[i][j][v];
      }
    }
}

// ---------------- softmax over K (rows of S); A1 in bf16 [l][k] ----------------
__global__ __launch_bounds__(256) void k_softmax_row(const float* __restrict__ S,
                                                     __bf16* __restrict__ A,
                                                     int Kd) {
  __shared__ float red[256];
  int t = threadIdx.x;
  const float* row = S + (size_t)blockIdx.x * Kd;
  float m = -3.4e38f;
#pragma unroll
  for (int j = 0; j < 4; ++j) m = fmaxf(m, row[t + j * 256]);
  red[t] = m;
  __syncthreads();
  for (int s = 128; s > 0; s >>= 1) {
    if (t < s) red[t] = fmaxf(red[t], red[t + s]);
    __syncthreads();
  }
  m = red[0];
  __syncthreads();
  float vals[4], sum = 0.f;
#pragma unroll
  for (int j = 0; j < 4; ++j) {
    vals[j] = __expf(row[t + j * 256] - m);
    sum += vals[j];
  }
  red[t] = sum;
  __syncthreads();
  for (int s = 128; s > 0; s >>= 1) {
    if (t < s) red[t] += red[t + s];
    __syncthreads();
  }
  float inv = 1.0f / red[0];
  __bf16* out = A + (size_t)blockIdx.x * Kd;
#pragma unroll
  for (int j = 0; j < 4; ++j) out[t + j * 256] = f2bf(vals[j] * inv);
}

// ------ softmax over L (columns of S); writes A2 TRANSPOSED: bf16 [n][k][l] ------
__global__ __launch_bounds__(256) void k_softmax_col(const float* __restrict__ S,
                                                     __bf16* __restrict__ A2t,
                                                     int L, int Kd) {
  __shared__ float red[256];
  int t = threadIdx.x;
  int n = blockIdx.x / Kd, k = blockIdx.x % Kd;
  const float* base = S + (size_t)n * L * Kd + k;
  float m = -3.4e38f;
#pragma unroll
  for (int j = 0; j < 4; ++j) m = fmaxf(m, base[(size_t)(t + j * 256) * Kd]);
  red[t] = m;
  __syncthreads();
  for (int s = 128; s > 0; s >>= 1) {
    if (t < s) red[t] = fmaxf(red[t], red[t + s]);
    __syncthreads();
  }
  m = red[0];
  __syncthreads();
  float vals[4], sum = 0.f;
#pragma unroll
  for (int j = 0; j < 4; ++j) {
    vals[j] = __expf(base[(size_t)(t + j * 256) * Kd] - m);
    sum += vals[j];
  }
  red[t] = sum;
  __syncthreads();
  for (int s = 128; s > 0; s >>= 1) {
    if (t < s) red[t] += red[t + s];
    __syncthreads();
  }
  float inv = 1.0f / red[0];
  __bf16* out = A2t + (size_t)n * L * Kd + (size_t)k * L;  // row k of [k][l]
#pragma unroll
  for (int j = 0; j < 4; ++j) out[t + j * 256] = f2bf(vals[j] * inv);  // coalesced
}

extern "C" void kernel_launch(void* const* d_in, const int* in_sizes, int n_in,
                              void* d_out, int out_size, void* d_ws, size_t ws_size,
                              hipStream_t stream) {
  (void)in_sizes; (void)n_in; (void)out_size; (void)ws_size;
  constexpr int N = 16, L = 1024, K = 1024, D = 1024;
  const float* a1 = (const float*)d_in[0];
  const float* a2 = (const float*)d_in[1];
  const float* Gw = (const float*)d_in[2];
  const float* Gb = (const float*)d_in[3];

  const size_t szA1 = (size_t)N * L * D;  // 16M elems
  const size_t szA2 = (size_t)N * K * D;
  const size_t szGw = (size_t)D * D;
  const size_t szS  = (size_t)N * L * K;

  char* w = (char*)d_ws;
  __bf16* a1b  = (__bf16*)w; w += szA1 * 2;  // [n][l][d]
  __bf16* a2b  = (__bf16*)w; w += szA2 * 2;  // [n][k][d]
  __bf16* Gwb  = (__bf16*)w; w += szGw * 2;  // [e][d]
  __bf16* a2pb = (__bf16*)w; w += szA2 * 2;  // [n][k][e]
  __bf16* a1t  = (__bf16*)w; w += szA1 * 2;  // [n][d][l]
  __bf16* a2t  = (__bf16*)w; w += szA2 * 2;  // [n][d][k]
  float*  Sf   = (float*) w; w += szS  * 4;  // [n][l][k]
  __bf16* A1b  = (__bf16*)w; w += szS  * 2;  // [n][l][k]
  __bf16* A2t  = (__bf16*)w;                 // [n][k][l]

  float* M1 = (float*)d_out;
  float* M2 = M1 + szA1;

  // 1) fp32 -> bf16 conversions (+ per-batch transposed copies for M1/M2 operands)
  k_cvt<<<(unsigned)(szA1 / 1024), 256, 0, stream>>>(a1, a1b, szA1);
  k_cvt<<<(unsigned)(szA2 / 1024), 256, 0, stream>>>(a2, a2b, szA2);
  k_cvt<<<(unsigned)(szGw / 1024), 256, 0, stream>>>(Gw, Gwb, szGw);
  dim3 gT(D / 32, L / 32, N);
  k_cvt_t<<<gT, 256, 0, stream>>>(a1, a1t, L, D);
  dim3 gT2(D / 32, K / 32, N);
  k_cvt_t<<<gT2, 256, 0, stream>>>(a2, a2t, K, D);

  // 2) a2p = a2 @ Gw^T + b   (rows N*K, cols D, reduce D)
  dim3 gLin(D / 256, (N * K) / 64, 1);
  k_linear<<<gLin, 256, 0, stream>>>(a2b, Gwb, Gb, a2pb, D);

  // 3) S = a1 @ a2p^T   (rows L, cols K, reduce D)
  dim3 gS(K / 256, L / 64, N);
  k_gemm_f32<<<gS, 256, 0, stream>>>(a1b, D, (size_t)L * D,
                                     a2pb, D, (size_t)K * D,
                                     Sf, K, (size_t)L * K, D);

  // 4) softmaxes: A1 = softmax_K(S) in [l][k]; A2^T = softmax_L(S) in [k][l]
  k_softmax_row<<<N * L, 256, 0, stream>>>(Sf, A1b, K);
  k_softmax_col<<<N * K, 256, 0, stream>>>(Sf, A2t, L, K);

  // 5) M1 = A1 @ a2   (rows L, cols D, reduce K; B from a2t [d][k])
  dim3 gM1(D / 256, L / 64, N);
  k_gemm_f32<<<gM1, 256, 0, stream>>>(A1b, K, (size_t)L * K,
                                      a2t, K, (size_t)K * D,
                                      M1, D, (size_t)L * D, K);

  // 6) M2 = A2^T @ a1  (rows K, cols D, reduce L; A from A2t [k][l], B from a1t [d][l])
  dim3 gM2(D / 256, K / 64, N);
  k_gemm_f32<<<gM2, 256, 0, stream>>>(A2t, L, (size_t)L * K,
                                      a1t, L, (size_t)L * D,
                                      M2, D, (size_t)K * D, L);
}
